// MindPalace_45775761441419
// MI455X (gfx1250) — matrix operand order repaired and checked by reading, not compiled, not run on hardware
//
#include <hip/hip_runtime.h>
#include <hip/hip_bf16.h>

typedef __attribute__((ext_vector_type(16))) _Float16 v16h;
typedef __attribute__((ext_vector_type(8)))  _Float16 v8h;
typedef __attribute__((ext_vector_type(4)))  _Float16 v4h;
typedef __attribute__((ext_vector_type(8)))  float    v8f;
typedef __attribute__((ext_vector_type(4)))  unsigned int v4u;
typedef __attribute__((ext_vector_type(8)))  int      v8i;
typedef __attribute__((ext_vector_type(4)))  int      v4i;

#define Bk   4
#define Tk   1024
#define Dk   1024
#define Hk   16
#define DHk  64
#define DFFk 4096
#define Rk   8

__device__ __forceinline__ float gelu_exact(float x) {
  return 0.5f * x * (1.0f + erff(x * 0.70710678118654752f));
}
__device__ __forceinline__ float sigm(float x) { return 1.0f / (1.0f + __expf(-x)); }

// A-fragment loader: lane L holds M=L&15, K = koff+{0..15}
// (koff = (L>>4)*16). Two 16B LDS reads.
__device__ __forceinline__ v16h frag_a(const _Float16* p) {
  v8h lo = *(const v8h*)p;
  v8h hi = *(const v8h*)(p + 8);
  v16h r;
#pragma unroll
  for (int i = 0; i < 8; ++i) { r[i] = lo[i]; r[i + 8] = hi[i]; }
  return r;
}

// ---------------------------------------------------------------------------
// TDM probe: NULL-descriptor tensor_load_to_lds (D#.count==0 -> architectural
// no-op) + s_wait_tensorcnt. Exercises the CDNA5 tensor DMA path safely.
// ---------------------------------------------------------------------------
__global__ void tdm_probe_k() {
#if defined(__gfx1250__) && __has_builtin(__builtin_amdgcn_tensor_load_to_lds)
  if (threadIdx.x == 0) {
    v4u g0 = {}; v8i g1 = {}; v4i g2 = {}; v4i g3 = {};
#if __clang_major__ >= 23
    v8i g4 = {};
    __builtin_amdgcn_tensor_load_to_lds(g0, g1, g2, g3, g4, 0);
#else
    __builtin_amdgcn_tensor_load_to_lds(g0, g1, g2, g3, 0);
#endif
  }
#endif
#if __has_builtin(__builtin_amdgcn_s_wait_tensorcnt)
  __builtin_amdgcn_s_wait_tensorcnt(0);
#endif
}

// ---------------------------------------------------------------------------
// Weight transpose+convert: W[K,N] f32 (room-indexed) -> Wt[N,K] f16.
// ---------------------------------------------------------------------------
__global__ __launch_bounds__(256) void wtrans_k(
    const float* __restrict__ Wb, _Float16* __restrict__ Wt,
    int K, int N, const int* __restrict__ idx, int hop, long wstride)
{
  __shared__ float tile[32][33];
  const float* W = Wb + (long)idx[hop] * wstride;
  const int n0 = blockIdx.x * 32, k0 = blockIdx.y * 32;
  const int tx = threadIdx.x & 31, ty = threadIdx.x >> 5;
#pragma unroll
  for (int i = 0; i < 4; ++i)
    tile[ty + i * 8][tx] = W[(long)(k0 + ty + i * 8) * N + n0 + tx];
  __syncthreads();
#pragma unroll
  for (int i = 0; i < 4; ++i)
    Wt[(long)(n0 + ty + i * 8) * K + k0 + tx] = (_Float16)tile[tx][ty + i * 8];
}

// ---------------------------------------------------------------------------
// WMMA GEMM, f16 in / f32 accumulate. A[M,K] f16 row-major, Wt[N,K] f16
// (pre-transposed). Block tile 128x64, K-tile 32, 256 threads (8 waves).
// Register-staged software pipeline + double-buffered LDS: global loads for
// tile kt+1 are issued before the WMMAs of tile kt, so s_wait_loadcnt lands
// after the matrix work. One barrier per K-tile, branch-free staging.
// ---------------------------------------------------------------------------
__global__ __launch_bounds__(256) void gemm_wmma_k(
    const _Float16* __restrict__ A, const _Float16* __restrict__ Wt,
    const float* __restrict__ Bb, const float* __restrict__ Res,
    float* __restrict__ Cf, _Float16* __restrict__ Ch,
    int M, int N, int K,
    const int* __restrict__ idx, int hop, long bstride, int act)
{
  __shared__ _Float16 lsA[2][128 * 40];   // [m][k], stride 40 halves
  __shared__ _Float16 lsB[2][64 * 40];    // [n][k], stride 40 halves
  (void)M;

  const float* Bi = Bb + (long)(idx ? idx[hop] : 0) * bstride;
  const int tid  = threadIdx.x;
  const int wave = tid >> 5;
  const int lane = tid & 31;
  const int m0 = blockIdx.y * 128;
  const int n0 = blockIdx.x * 64;
  const int nkt = K >> 5;

  // Per-thread staging coordinates (128x32 A tile = 512 v8h, 2/thread;
  // 64x32 B tile = 256 v8h, 1/thread).
  const int arow0 = tid >> 2;            // rows 0..63
  const int arow1 = arow0 + 64;          // rows 64..127
  const int ac    = tid & 3;
  const long abase0 = (long)(m0 + arow0) * K + ac * 8;
  const long abase1 = (long)(m0 + arow1) * K + ac * 8;
  const long bbase  = (long)(n0 + arow0) * K + ac * 8;

  v8h ra0 = *(const v8h*)&A[abase0];
  v8h ra1 = *(const v8h*)&A[abase1];
  v8h rb  = *(const v8h*)&Wt[bbase];

  v8f acc[4] = {};
  const int koff = (lane >> 4) * 16;
  for (int kt = 0; kt < nkt; ++kt) {
    const int buf = kt & 1;
    // Commit staged registers (loadcnt wait lands here, after prior compute).
    *(v8h*)&lsA[buf][arow0 * 40 + ac * 8] = ra0;
    *(v8h*)&lsA[buf][arow1 * 40 + ac * 8] = ra1;
    *(v8h*)&lsB[buf][arow0 * 40 + ac * 8] = rb;
    __syncthreads();
    // Issue next tile's loads; they retire during this tile's WMMAs.
    const long ko = (long)((kt + 1 < nkt) ? kt + 1 : kt) * 32;
    ra0 = *(const v8h*)&A[abase0 + ko];
    ra1 = *(const v8h*)&A[abase1 + ko];
    rb  = *(const v8h*)&Wt[bbase + ko];
    const long kp = (long)((kt + 2 < nkt) ? kt + 2 : nkt - 1) * 32;
    __builtin_prefetch(&A[abase0 + kp], 0, 1);
    __builtin_prefetch(&Wt[bbase + kp], 0, 1);

    v16h af = frag_a(&lsA[buf][(wave * 16 + (lane & 15)) * 40 + koff]);
#pragma unroll
    for (int j = 0; j < 4; ++j) {
      v16h bf = *(const v16h*)&lsB[buf][(j * 16 + (lane & 15)) * 40 + ((lane >> 4) << 4)];
      acc[j] = __builtin_amdgcn_wmma_f32_16x16x32_f16(false, af, false, bf,
                                                      (short)0, acc[j], false, false);
    }
  }

#pragma unroll
  for (int j = 0; j < 4; ++j) {
    const int n = n0 + j * 16 + (lane & 15);
    const float bias = Bi[n];
#pragma unroll
    for (int rr = 0; rr < 8; ++rr) {
      const int m = m0 + wave * 16 + (lane >> 4) * 8 + rr;
      float v = acc[j][rr] + bias;
      if (act) v = gelu_exact(v);
      if (Res) v += Res[(long)m * N + n];
      if (Ch) Ch[(long)m * N + n] = (_Float16)v;
      else    Cf[(long)m * N + n] = v;
    }
  }
}

// ---------------------------------------------------------------------------
// Flash attention (causal), f16 in / f16 out, WMMA for QK^T and PV.
// Grid: (T/64, H, B); 128 threads = 4 waves; wave owns 16 query rows.
// K/V tiles register-staged one iteration ahead.
// ---------------------------------------------------------------------------
__global__ __launch_bounds__(128) void flash_attn_k(
    const _Float16* __restrict__ Q, const _Float16* __restrict__ Kg,
    const _Float16* __restrict__ Vg, _Float16* __restrict__ O)
{
  __shared__ _Float16 Kls[64 * 72];       // [key][dh] (also Q staging)
  __shared__ _Float16 Vls[64 * 72];       // [dh][key]
  __shared__ _Float16 Pls[4][16 * 72];    // per-wave P [m][key]

  const int tid  = threadIdx.x;
  const int wave = tid >> 5;
  const int lane = tid & 31;
  const int qt = blockIdx.x, h = blockIdx.y, b = blockIdx.z;
  const long headoff = (long)b * Tk * Dk + h * DHk;
  const int koff = (lane >> 4) * 16;

  // Stage Q tile through Kls; grab two A-fragments (dh chunks of 32)
#pragma unroll
  for (int i = 0; i < 4; ++i) {
    int id = tid + 128 * i;
    int row = id >> 3, c = id & 7;
    *(v8h*)&Kls[row * 72 + c * 8] =
        *(const v8h*)&Q[headoff + (long)(qt * 64 + row) * Dk + c * 8];
  }
  __syncthreads();
  v16h qf[2];
#pragma unroll
  for (int c = 0; c < 2; ++c)
    qf[c] = frag_a(&Kls[(wave * 16 + (lane & 15)) * 72 + c * 32 + koff]);

  float mrow[8], lrow[8];
  v8f oacc[4] = {};
#pragma unroll
  for (int rr = 0; rr < 8; ++rr) { mrow[rr] = -1e30f; lrow[rr] = 0.0f; }

  // Register-staged K/V tiles
  v8h rk[4], rv[4];
#pragma unroll
  for (int i = 0; i < 4; ++i) {
    int id = tid + 128 * i;
    int row = id >> 3, c = id & 7;
    rk[i] = *(const v8h*)&Kg[headoff + (long)row * Dk + c * 8];
    rv[i] = *(const v8h*)&Vg[headoff + (long)row * Dk + c * 8];
  }

  for (int kt = 0; kt <= qt; ++kt) {
    __syncthreads();                    // prior tile's LDS reads done
#pragma unroll
    for (int i = 0; i < 4; ++i) {
      int id = tid + 128 * i;
      int row = id >> 3, c = id & 7;
      *(v8h*)&Kls[row * 72 + c * 8] = rk[i];
#pragma unroll
      for (int j = 0; j < 8; ++j) Vls[(c * 8 + j) * 72 + row] = rv[i][j];
    }
    __syncthreads();
    // Issue next tile's loads; retire during this tile's compute.
    const int ktn = (kt + 1 <= qt) ? kt + 1 : kt;
#pragma unroll
    for (int i = 0; i < 4; ++i) {
      int id = tid + 128 * i;
      int row = id >> 3, c = id & 7;
      rk[i] = *(const v8h*)&Kg[headoff + (long)(ktn * 64 + row) * Dk + c * 8];
      rv[i] = *(const v8h*)&Vg[headoff + (long)(ktn * 64 + row) * Dk + c * 8];
    }

    // S = Q * K^T
    v8f sacc[4] = {};
#pragma unroll
    for (int c = 0; c < 2; ++c) {
#pragma unroll
      for (int j = 0; j < 4; ++j) {
        v16h bf = *(const v16h*)&Kls[(j * 16 + (lane & 15)) * 72 + c * 32 + ((lane >> 4) << 4)];
        sacc[j] = __builtin_amdgcn_wmma_f32_16x16x32_f16(false, qf[c], false, bf,
                                                         (short)0, sacc[j], false, false);
      }
    }

    // scale + causal mask + online softmax
    const bool diag = (kt == qt);
    float pmax[8];
#pragma unroll
    for (int rr = 0; rr < 8; ++rr) pmax[rr] = -1e30f;
#pragma unroll
    for (int j = 0; j < 4; ++j) {
#pragma unroll
      for (int rr = 0; rr < 8; ++rr) {
        float s = sacc[j][rr] * 0.125f;
        if (diag) {
          int mg = wave * 16 + (lane >> 4) * 8 + rr;
          int ng = j * 16 + (lane & 15);
          if (ng > mg) s = -1e30f;
        }
        sacc[j][rr] = s;
        pmax[rr] = fmaxf(pmax[rr], s);
      }
    }
#pragma unroll
    for (int rr = 0; rr < 8; ++rr) {
      float v = pmax[rr];
#pragma unroll
      for (int msk = 1; msk < 16; msk <<= 1) v = fmaxf(v, __shfl_xor(v, msk, 32));
      float mnew  = fmaxf(mrow[rr], v);
      float alpha = __expf(mrow[rr] - mnew);
      mrow[rr] = mnew;
      lrow[rr] *= alpha;
#pragma unroll
      for (int j = 0; j < 4; ++j) oacc[j][rr] *= alpha;
      float lpart = 0.0f;
#pragma unroll
      for (int j = 0; j < 4; ++j) {
        float p = __expf(sacc[j][rr] - mnew);
        lpart += p;
        Pls[wave][(rr + (lane >> 4) * 8) * 72 + j * 16 + (lane & 15)] = (_Float16)p;
      }
#pragma unroll
      for (int msk = 1; msk < 16; msk <<= 1) lpart += __shfl_xor(lpart, msk, 32);
      lrow[rr] += lpart;
    }
    __syncthreads();   // cross-lane P stores visible

    // O += P * V
#pragma unroll
    for (int c = 0; c < 2; ++c) {
      v16h pf = frag_a(&Pls[wave][(lane & 15) * 72 + c * 32 + koff]);
#pragma unroll
      for (int j = 0; j < 4; ++j) {
        v16h vf = *(const v16h*)&Vls[(j * 16 + (lane & 15)) * 72 + c * 32 + ((lane >> 4) << 4)];
        oacc[j] = __builtin_amdgcn_wmma_f32_16x16x32_f16(false, pf, false, vf,
                                                         (short)0, oacc[j], false, false);
      }
    }
  }

#pragma unroll
  for (int j = 0; j < 4; ++j) {
    int dh = j * 16 + (lane & 15);
#pragma unroll
    for (int rr = 0; rr < 8; ++rr) {
      int t = qt * 64 + wave * 16 + (lane >> 4) * 8 + rr;
      O[headoff + (long)t * Dk + dh] = (_Float16)(oacc[j][rr] / lrow[rr]);
    }
  }
}

// ---------------------------------------------------------------------------
// LayerNorm over D=1024 (one block per row). Optional per-batch gate on input
// (xg_out gets the gated f32 input); normalized output stored f16.
// ---------------------------------------------------------------------------
__global__ __launch_bounds__(256) void ln_k(
    const float* __restrict__ x, const float* __restrict__ gate,
    const float* __restrict__ gvec, const float* __restrict__ bvec,
    const int* __restrict__ idx, int hop,
    float* __restrict__ xg_out, _Float16* __restrict__ out)
{
  __shared__ float red[256];
  const int row = blockIdx.x;           // b*T + t
  const int b   = row >> 10;            // T == 1024
  const int r   = idx[hop];
  const float g = gate ? gate[b] : 1.0f;
  const long base = (long)row * Dk;
  float v[4];
  float s = 0.0f;
#pragma unroll
  for (int i = 0; i < 4; ++i) {
    int d = threadIdx.x + 256 * i;
    v[i] = x[base + d] * g;
    s += v[i];
    if (xg_out) xg_out[base + d] = v[i];
  }
  red[threadIdx.x] = s; __syncthreads();
  for (int st = 128; st > 0; st >>= 1) {
    if (threadIdx.x < st) red[threadIdx.x] += red[threadIdx.x + st];
    __syncthreads();
  }
  float mean = red[0] * (1.0f / Dk);
  __syncthreads();
  float vs = 0.0f;
#pragma unroll
  for (int i = 0; i < 4; ++i) { float dv = v[i] - mean; vs += dv * dv; }
  red[threadIdx.x] = vs; __syncthreads();
  for (int st = 128; st > 0; st >>= 1) {
    if (threadIdx.x < st) red[threadIdx.x] += red[threadIdx.x + st];
    __syncthreads();
  }
  float inv = rsqrtf(red[0] * (1.0f / Dk) + 1e-5f);
#pragma unroll
  for (int i = 0; i < 4; ++i) {
    int d = threadIdx.x + 256 * i;
    out[base + d] = (_Float16)((v[i] - mean) * inv * gvec[r * Dk + d] + bvec[r * Dk + d]);
  }
}

// Mean over T: out[b*D+d] = mean_t x[b][t][d]
__global__ __launch_bounds__(256) void meant_k(const float* __restrict__ x,
                                               float* __restrict__ out)
{
  int i = blockIdx.x * 256 + threadIdx.x;
  int b = i >> 10, d = i & 1023;
  float s = 0.0f;
  for (int t = 0; t < Tk; ++t) s += x[(long)(b * Tk + t) * Dk + d];
  out[i] = s * (1.0f / Tk);
}

// ctx = ctx0 @ ctx_w + ctx_b
__global__ __launch_bounds__(256) void linD_k(const float* __restrict__ in,
    const float* __restrict__ w, const float* __restrict__ bias, float* __restrict__ out)
{
  int i = blockIdx.x * 256 + threadIdx.x;
  int b = i >> 10, n = i & 1023;
  float acc = bias[n];
  for (int k = 0; k < Dk; ++k) acc += in[b * Dk + k] * w[(long)k * Dk + n];
  out[i] = acc;
}

// Router: warp/adj/direct/boost/gate_scores (single block)
__global__ __launch_bounds__(256) void router_k(
    const float* __restrict__ ctx, const float* __restrict__ summaries,
    const float* __restrict__ base_adj, const float* __restrict__ warp_w,
    const float* __restrict__ warp_b, const float* __restrict__ gate_w,
    const float* __restrict__ gate_b,
    float* __restrict__ gs_ws, float* __restrict__ gs_out, float* __restrict__ adj_out)
{
  __shared__ float warp[Bk][64];
  __shared__ float adjl[Bk][64];
  __shared__ float dir[Bk][8];
  const int t = threadIdx.x;
  {
    int b = t >> 6, j = t & 63;
    float acc = warp_b[j];
    for (int k = 0; k < Dk; ++k) acc += ctx[b * Dk + k] * warp_w[(long)k * 64 + j];
    warp[b][j] = 0.1f * acc;
  }
  __syncthreads();
  if (t < 32) {
    int b = t >> 3, i = t & 7;
    float vrow[8]; float mx = -1e30f;
    for (int j = 0; j < 8; ++j) {
      vrow[j] = base_adj[i * 8 + j] + warp[b][i * 8 + j];
      mx = fmaxf(mx, vrow[j]);
    }
    float ssum = 0.0f;
    for (int j = 0; j < 8; ++j) { vrow[j] = __expf(vrow[j] - mx); ssum += vrow[j]; }
    for (int j = 0; j < 8; ++j) {
      float a = vrow[j] / ssum;
      adjl[b][i * 8 + j] = a;
      adj_out[b * 64 + i * 8 + j] = a;
    }
    float ds = 0.0f;
    for (int k = 0; k < Dk; ++k) ds += ctx[b * Dk + k] * summaries[i * Dk + k];
    dir[b][i] = ds;
  }
  __syncthreads();
  if (t < 32) {
    int b = t >> 3, rr = t & 7;
    float boost = 0.0f;
    for (int j = 0; j < 8; ++j) boost += adjl[b][rr * 8 + j] * dir[b][j];
    float g = gate_b[rr];
    for (int k = 0; k < Dk; ++k) g += ctx[b * Dk + k] * gate_w[(long)k * 8 + rr];
    float v = sigm((g + boost) * 0.5f);
    gs_ws[b * 8 + rr] = v;
    gs_out[b * 8 + rr] = v;
  }
}

// top-3 of batch-mean gate scores; writes MAX_HOPS scalar
__global__ void topk_k(const float* __restrict__ gs, int* __restrict__ idx,
                       float* __restrict__ hops_out)
{
  if (threadIdx.x == 0 && blockIdx.x == 0) {
    float m[8];
    for (int rr = 0; rr < 8; ++rr)
      m[rr] = 0.25f * (gs[rr] + gs[8 + rr] + gs[16 + rr] + gs[24 + rr]);
    bool used[8] = {};
    for (int h = 0; h < 3; ++h) {
      float best = -1e30f; int bi = 0;
      for (int rr = 0; rr < 8; ++rr)
        if (!used[rr] && m[rr] > best) { best = m[rr]; bi = rr; }
      used[bi] = true;
      idx[h] = bi;
    }
    *hops_out = 3.0f;
  }
}

// Per-room gate: sigmoid(gelu(ctx @ g1 + b1) @ g2 + b2); gm = batch mean
__global__ __launch_bounds__(256) void room_gate_k(
    const float* __restrict__ ctxr, const float* __restrict__ g1w,
    const float* __restrict__ g1b, const float* __restrict__ g2w,
    const float* __restrict__ g2b, const int* __restrict__ idx, int hop,
    float* __restrict__ gate, float* __restrict__ gm)
{
  __shared__ float hbuf[Bk][256];
  const int r = idx[hop];
  const int j = threadIdx.x;
  const float* w1 = g1w + (long)r * Dk * 256;
  const float w2  = g2w[r * 256 + j];
#pragma unroll
  for (int b = 0; b < Bk; ++b) {
    float acc = g1b[r * 256 + j];
    for (int k = 0; k < Dk; ++k) acc += ctxr[b * Dk + k] * w1[(long)k * 256 + j];
    hbuf[b][j] = gelu_exact(acc) * w2;
  }
  __syncthreads();
  for (int s = 128; s > 0; s >>= 1) {
    if (j < s) {
#pragma unroll
      for (int b = 0; b < Bk; ++b) hbuf[b][j] += hbuf[b][j + s];
    }
    __syncthreads();
  }
  if (j == 0) {
    float s = 0.0f;
#pragma unroll
    for (int b = 0; b < Bk; ++b) {
      float gv = sigm(hbuf[b][0] + g2b[r]);
      gate[b] = gv;
      s += gv;
    }
    *gm = s * (1.0f / Bk);
  }
}

// cur = cur + gm * (xo - cur)
__global__ __launch_bounds__(256) void mix_k(float* __restrict__ cur,
    const float* __restrict__ xo, const float* __restrict__ gm)
{
  long i = (long)blockIdx.x * 1024 + threadIdx.x * 4;
  float g = *gm;
  float4 c = *(float4*)&cur[i];
  float4 o = *(const float4*)&xo[i];
  c.x += g * (o.x - c.x); c.y += g * (o.y - c.y);
  c.z += g * (o.z - c.z); c.w += g * (o.w - c.w);
  *(float4*)&cur[i] = c;
}

__global__ __launch_bounds__(256) void copy_k(float* __restrict__ dst,
                                              const float* __restrict__ src)
{
  long i = (long)blockIdx.x * 1024 + threadIdx.x * 4;
  *(float4*)&dst[i] = *(const float4*)&src[i];
}

// ---------------------------------------------------------------------------
extern "C" void kernel_launch(void* const* d_in, const int* in_sizes, int n_in,
                              void* d_out, int out_size, void* d_ws, size_t ws_size,
                              hipStream_t stream)
{
  (void)in_sizes; (void)n_in; (void)out_size; (void)ws_size;
  const float* x         = (const float*)d_in[0];
  const float* summaries = (const float*)d_in[1];
  const float* ctx_w     = (const float*)d_in[2];
  const float* ctx_b     = (const float*)d_in[3];
  const float* base_adj  = (const float*)d_in[4];
  const float* warp_w    = (const float*)d_in[5];
  const float* warp_b    = (const float*)d_in[6];
  const float* gate_w    = (const float*)d_in[7];
  const float* gate_b    = (const float*)d_in[8];
  const float* g1_w  = (const float*)d_in[9];
  const float* g1_b  = (const float*)d_in[10];
  const float* g2_w  = (const float*)d_in[11];
  const float* g2_b  = (const float*)d_in[12];
  const float* ln1_g = (const float*)d_in[13];
  const float* ln1_b = (const float*)d_in[14];
  const float* wq = (const float*)d_in[15];
  const float* bq = (const float*)d_in[16];
  const float* wk = (const float*)d_in[17];
  const float* bkp = (const float*)d_in[18];
  const float* wv = (const float*)d_in[19];
  const float* bv = (const float*)d_in[20];
  const float* wo = (const float*)d_in[21];
  const float* bo = (const float*)d_in[22];
  const float* ln2_g = (const float*)d_in[23];
  const float* ln2_b = (const float*)d_in[24];
  const float* ff1_w = (const float*)d_in[25];
  const float* ff1_b = (const float*)d_in[26];
  const float* ff2_w = (const float*)d_in[27];
  const float* ff2_b = (const float*)d_in[28];
  float* out = (float*)d_out;

  const long BTD = (long)Bk * Tk * Dk;          // 4,194,304
  float* ws    = (float*)d_ws;
  float* ctx0  = ws;                            // B*D
  float* ctxv  = ws + 4096;
  float* gsw   = ws + 8192;                     // B*R
  float* gatev = ws + 8224;                     // B
  float* gmp   = ws + 8228;                     // 1
  int*   idxp  = (int*)(ws + 8232);             // MAX_HOPS ints
  float* cur   = ws + 16384;                    // f32 state
  float* xg    = cur + BTD;                     // f32 gated input (residual)
  float* xo    = xg  + BTD;                     // f32 block output
  _Float16* h0   = (_Float16*)(xo + BTD);       // f16 region
  _Float16* nrmh = h0;                          // BTD
  _Float16* qh   = nrmh + BTD;
  _Float16* kh   = qh + BTD;
  _Float16* vh   = kh + BTD;
  _Float16* oh   = vh + BTD;
  _Float16* hffh = oh + BTD;                    // BTD * 4
  _Float16* wt   = hffh + BTD * 4;              // up to D*DFF halves

  // ---- TDM/TENSORcnt probe (architectural no-op) ----
  tdm_probe_k<<<1, 32, 0, stream>>>();

  // ---- Router ----
  meant_k<<<16, 256, 0, stream>>>(x, ctx0);
  linD_k <<<16, 256, 0, stream>>>(ctx0, ctx_w, ctx_b, ctxv);
  router_k<<<1, 256, 0, stream>>>(ctxv, summaries, base_adj, warp_w, warp_b,
                                  gate_w, gate_b, gsw, out + BTD, out + BTD + 32);
  topk_k<<<1, 1, 0, stream>>>(gsw, idxp, out + BTD + 32 + 256);

  copy_k<<<(int)(BTD / 1024), 256, 0, stream>>>(cur, x);

  const int MB = Bk * Tk;                       // 4096 rows
  const dim3 gq  (Dk / 64,   MB / 128);         // N=1024 GEMMs
  const dim3 gf1 (DFFk / 64, MB / 128);         // N=4096 GEMM
  const dim3 ga  (Tk / 64, Hk, Bk);
  const dim3 tDD (Dk / 32,   Dk / 32);          // 1024x1024 transpose
  const dim3 tDF (DFFk / 32, Dk / 32);          // ff1: K=D,   N=DFF
  const dim3 tFD (Dk / 32,   DFFk / 32);        // ff2: K=DFF, N=D

  for (int hop = 0; hop < 3; ++hop) {
    // room gate
    meant_k<<<16, 256, 0, stream>>>(cur, ctx0);
    room_gate_k<<<1, 256, 0, stream>>>(ctx0, g1_w, g1_b, g2_w, g2_b,
                                       idxp, hop, gatev, gmp);
    // xg = cur * gate ; nrmh = LN1(xg) (f16)
    ln_k<<<MB, 256, 0, stream>>>(cur, gatev, ln1_g, ln1_b, idxp, hop, xg, nrmh);

    // QKV projections (f16 out)
    wtrans_k<<<tDD, 256, 0, stream>>>(wq, wt, Dk, Dk, idxp, hop, (long)Dk * Dk);
    gemm_wmma_k<<<gq, 256, 0, stream>>>(nrmh, wt, bq, nullptr, nullptr, qh,
                                        MB, Dk, Dk, idxp, hop, Dk, 0);
    wtrans_k<<<tDD, 256, 0, stream>>>(wk, wt, Dk, Dk, idxp, hop, (long)Dk * Dk);
    gemm_wmma_k<<<gq, 256, 0, stream>>>(nrmh, wt, bkp, nullptr, nullptr, kh,
                                        MB, Dk, Dk, idxp, hop, Dk, 0);
    wtrans_k<<<tDD, 256, 0, stream>>>(wv, wt, Dk, Dk, idxp, hop, (long)Dk * Dk);
    gemm_wmma_k<<<gq, 256, 0, stream>>>(nrmh, wt, bv, nullptr, nullptr, vh,
                                        MB, Dk, Dk, idxp, hop, Dk, 0);
    // causal MHA (f16 out)
    flash_attn_k<<<ga, 128, 0, stream>>>(qh, kh, vh, oh);
    // xo = xg + o @ wo + bo (f32 out)
    wtrans_k<<<tDD, 256, 0, stream>>>(wo, wt, Dk, Dk, idxp, hop, (long)Dk * Dk);
    gemm_wmma_k<<<gq, 256, 0, stream>>>(oh, wt, bo, xg, xo, nullptr,
                                        MB, Dk, Dk, idxp, hop, Dk, 0);
    // nrmh = LN2(xo) (f16)
    ln_k<<<MB, 256, 0, stream>>>(xo, nullptr, ln2_g, ln2_b, idxp, hop, nullptr, nrmh);
    // hffh = gelu(nrmh @ ff1 + b1) (f16 out)
    wtrans_k<<<tDF, 256, 0, stream>>>(ff1_w, wt, Dk, DFFk, idxp, hop, (long)Dk * DFFk);
    gemm_wmma_k<<<gf1, 256, 0, stream>>>(nrmh, wt, ff1_b, nullptr, nullptr, hffh,
                                         MB, DFFk, Dk, idxp, hop, DFFk, 1);
    // xo = xo + hffh @ ff2 + b2 (f32 out, in-place residual)
    wtrans_k<<<tFD, 256, 0, stream>>>(ff2_w, wt, DFFk, Dk, idxp, hop, (long)DFFk * Dk);
    gemm_wmma_k<<<gq, 256, 0, stream>>>(hffh, wt, ff2_b, xo, xo, nullptr,
                                        MB, Dk, DFFk, idxp, hop, Dk, 0);
    // cur = cur + gm * (xo - cur)
    mix_k<<<(int)(BTD / 1024), 256, 0, stream>>>(cur, xo, gmp);
  }

  copy_k<<<(int)(BTD / 1024), 256, 0, stream>>>(out, cur);
}
